// MultiHeadAttention_56727928045677
// MI455X (gfx1250) — matrix operand order repaired and checked by reading, not compiled, not run on hardware
//
#include <hip/hip_runtime.h>
#include <hip/hip_bf16.h>

// MHA: B=2, N=2048, C=1024, H=16, D=64
#define SEQ   2048
#define EMB   1024
#define HEADS 16
#define HDIM  64
#define MROWS 4096   // B*N

typedef __attribute__((ext_vector_type(16))) __bf16 v16bf;
typedef __attribute__((ext_vector_type(8)))  float  v8f;

union FragB16 {
    uint4        q[2];
    unsigned int u[8];
    v16bf        v;
};

// --- f32 -> packed bf16 (RNE). Prefer v_cvt_pk_bf16_f32 when available. ---
#if __has_builtin(__builtin_amdgcn_cvt_pk_bf16_f32)
typedef __attribute__((ext_vector_type(2))) __bf16 v2bf;
__device__ __forceinline__ unsigned int pk2bf(float lo, float hi) {
    union { v2bf v; unsigned int u; } cvt;
    cvt.v = __builtin_amdgcn_cvt_pk_bf16_f32(lo, hi);
    return cvt.u;
}
#else
__device__ __forceinline__ unsigned int pk2bf(float lo, float hi) {
    unsigned int a = __float_as_uint(lo), b = __float_as_uint(hi);
    a += 0x7FFFu + ((a >> 16) & 1u);
    b += 0x7FFFu + ((b >> 16) & 1u);
    return (a >> 16) | (b & 0xFFFF0000u);
}
#endif
__device__ __forceinline__ unsigned short f2bf(float f) {
    return (unsigned short)pk2bf(f, f);
}
__device__ __forceinline__ v8f wmma_bf16(const FragB16& a, const FragB16& b, v8f c) {
    return __builtin_amdgcn_wmma_f32_16x16x32_bf16(
        false, a.v, false, b.v, (short)0, c, false, false);
}
// low 32 bits of a generic pointer into __shared__ == raw LDS byte offset
__device__ __forceinline__ unsigned lds_off(const void* p) {
    return (unsigned)(unsigned long long)p;
}
// per-lane async copy: LDS[lds+ofs .. +15] = MEM[gaddr+ofs .. +15]
#define ASYNC_CP16(lds, gaddr, ofs)                                             \
    asm volatile("global_load_async_to_lds_b128 %0, %1, off offset:" #ofs       \
                 :: "v"(lds), "v"(gaddr) : "memory")
#define ASYNC_WAIT() asm volatile("s_wait_asynccnt 0x0" ::: "memory")

// ---------------------------------------------------------------------------
// Kernel 1: fp32 -> bf16 elementwise convert
// ---------------------------------------------------------------------------
__global__ void cvt_f32_bf16(const float* __restrict__ in,
                             unsigned short* __restrict__ out, int n) {
    int i = (blockIdx.x * blockDim.x + threadIdx.x) * 4;
    if (i + 3 < n) {
        float4 f = *(const float4*)(in + i);
        uint2 o;
        o.x = pk2bf(f.x, f.y);
        o.y = pk2bf(f.z, f.w);
        *(uint2*)(out + i) = o;
    }
}

// ---------------------------------------------------------------------------
// Kernel 2: GEMM  Y[m,n] = sum_k A_bf16[m,k] * W_f32[k,n] + bias[n]
// MODE 0: store bf16 head-split [b,h,ns,d];  MODE 1: store f32 [m][1024]
// Block 256 threads (8 waves), tile 128x128, k-step 32, double-buffered LDS,
// A tile staged with GLOBAL_LOAD_ASYNC_TO_LDS_B128 (ASYNCcnt).
// ---------------------------------------------------------------------------
template <int MODE>
__global__ __launch_bounds__(256) void gemm_bf16(
    const unsigned short* __restrict__ A,
    const float* __restrict__ W,
    const float* __restrict__ bias,
    unsigned short* __restrict__ outQ,
    float* __restrict__ out32) {

    __shared__ __align__(16) unsigned short As[2][128 * 32];   // [m][k]
    __shared__ __align__(16) unsigned short BT[2][128 * 32];   // [n][k] transposed

    const int tid  = threadIdx.x;
    const int w    = tid >> 5;
    const int L    = tid & 31;
    const int half = L >> 4;
    const int l16  = L & 15;
    const int wm   = w >> 1;            // 0..3  (32-row strip)
    const int wn   = w & 1;             // 0..1  (64-col strip)
    const int m0   = blockIdx.x * 128;
    const int n0   = blockIdx.y * 128;

    v8f c[2][4];
    for (int i = 0; i < 2; ++i)
        for (int j = 0; j < 4; ++j)
            for (int e = 0; e < 8; ++e) c[i][j][e] = 0.0f;

    const int ar  = tid >> 1;           // A tile row 0..127
    const int ac  = (tid & 1) * 16;     // A tile col chunk (elems)
    const int bn  = tid & 127;          // W tile n 0..127
    const int bkg = tid >> 7;           // W tile k-half 0/1

    auto stageA = [&](int kk, int buf) {
        const unsigned short* ga = A + (size_t)(m0 + ar) * EMB + kk + ac;
        unsigned la = lds_off(&As[buf][ar * 32 + ac]);
        ASYNC_CP16(la, ga, 0);
        ASYNC_CP16(la, ga, 16);
    };
    auto stageW = [&](int kk, int buf) {
        const float* gw = W + (size_t)(kk + bkg * 16) * EMB + n0 + bn;
        unsigned t[8];
#pragma unroll
        for (int i = 0; i < 8; ++i)
            t[i] = pk2bf(gw[(2 * i) * EMB], gw[(2 * i + 1) * EMB]);
        uint4* lb = (uint4*)&BT[buf][bn * 32 + bkg * 16];
        lb[0] = make_uint4(t[0], t[1], t[2], t[3]);
        lb[1] = make_uint4(t[4], t[5], t[6], t[7]);
    };

    // prologue: stage tile 0
    stageA(0, 0);
    stageW(0, 0);
    ASYNC_WAIT();
    __syncthreads();

    for (int kk = 0; kk < EMB; kk += 32) {
        const int cur = (kk >> 5) & 1;
        const int nxt = cur ^ 1;
        if (kk + 32 < EMB) {            // prefetch next tile into other buffer
            stageA(kk + 32, nxt);
            stageW(kk + 32, nxt);
        }

        FragB16 af[2], bfr[4];
#pragma unroll
        for (int i = 0; i < 2; ++i) {
            const uint4* p = (const uint4*)&As[cur][(wm * 32 + i * 16 + l16) * 32];
            af[i].q[0] = p[half * 2];    // K = half*16 + {0..15}
            af[i].q[1] = p[half * 2 + 1];
        }
#pragma unroll
        for (int j = 0; j < 4; ++j) {
            const uint4* p = (const uint4*)&BT[cur][(wn * 64 + j * 16 + l16) * 32];
            bfr[j].q[0] = p[half * 2];   // K = half*16 + {0..15}
            bfr[j].q[1] = p[half * 2 + 1];
        }
#pragma unroll
        for (int i = 0; i < 2; ++i)
#pragma unroll
            for (int j = 0; j < 4; ++j)
                c[i][j] = wmma_bf16(af[i], bfr[j], c[i][j]);

        ASYNC_WAIT();
        __syncthreads();
    }

    // --- epilogue: bias + store (MODE is a template arg, no runtime branch) ---
#pragma unroll
    for (int j = 0; j < 4; ++j) {
        const int col = n0 + wn * 64 + j * 16 + l16;
        const float bval = bias[col];
#pragma unroll
        for (int i = 0; i < 2; ++i) {
#pragma unroll
            for (int v = 0; v < 8; ++v) {
                const int row = m0 + wm * 32 + i * 16 + half * 8 + v;
                const float val = c[i][j][v] + bval;
                if (MODE == 1) {
                    out32[(size_t)row * EMB + col] = val;
                } else {
                    const int bb = row >> 11;
                    const int ns = row & (SEQ - 1);
                    const int hh = col >> 6;
                    const int dd = col & (HDIM - 1);
                    outQ[(((size_t)(bb * HEADS + hh) * SEQ) + ns) * HDIM + dd] = f2bf(val);
                }
            }
        }
    }
}

// ---------------------------------------------------------------------------
// Kernel 3: attention per (b, h, 16-query strip). 2 waves / 64 threads.
// Phase 1: S[16][2048] = Q Kt via WMMA (f32 in LDS, each wave half the keys)
// Phase 2: softmax max/exp sweeps (float4); 1/sum deferred to phase 3
// Phase 3: ctx = attn @ V via WMMA, V tiles async-staged + double-buffered,
//          row normalization folded into the f32->bf16 fragment conversion
// ---------------------------------------------------------------------------
__global__ __launch_bounds__(64) void attn_kernel(
    const unsigned short* __restrict__ Q,
    const unsigned short* __restrict__ K,
    const unsigned short* __restrict__ V,
    unsigned short* __restrict__ ctx) {

    __shared__ __align__(16) float S[16 * SEQ];                 // 128 KB
    __shared__ __align__(16) unsigned short Vt[2][32 * HDIM];   // 2 x 4 KB
    __shared__ float redA[64], redB[64], invR[16];

    const int tid  = threadIdx.x;
    const int w    = tid >> 5;
    const int L    = tid & 31;
    const int half = L >> 4;
    const int l16  = L & 15;
    const int n0   = blockIdx.x * 16;
    const int h    = blockIdx.y;
    const int b    = blockIdx.z;
    const size_t headBase = (size_t)(b * HEADS + h) * SEQ * HDIM;

    // Q fragments: rows n0..n0+15, two k-steps over d=64
    FragB16 aQ[2];
    {
        const uint4* qp = (const uint4*)(Q + headBase + (size_t)(n0 + l16) * HDIM);
#pragma unroll
        for (int s = 0; s < 2; ++s) {
            aQ[s].q[0] = qp[s * 4 + half * 2];   // d = s*32 + half*16 ..
            aQ[s].q[1] = qp[s * 4 + half * 2 + 1];
        }
    }

    // ---- Phase 1: scores ----
#pragma unroll 4
    for (int mt = 0; mt < 64; ++mt) {
        const int m0k = (w * 64 + mt) * 16;
        const uint4* kp = (const uint4*)(K + headBase + (size_t)(m0k + l16) * HDIM);
        FragB16 bK[2];
#pragma unroll
        for (int s = 0; s < 2; ++s) {            // Kt fragment: d contiguous per lane
            bK[s].q[0] = kp[s * 4 + half * 2];
            bK[s].q[1] = kp[s * 4 + half * 2 + 1];
        }
        v8f c;
        for (int e = 0; e < 8; ++e) c[e] = 0.0f;
        c = wmma_bf16(aQ[0], bK[0], c);
        c = wmma_bf16(aQ[1], bK[1], c);
#pragma unroll
        for (int v = 0; v < 8; ++v)
            S[(half * 8 + v) * SEQ + m0k + l16] = c[v];
    }
    __syncthreads();

    // ---- Phase 2: softmax statistics (4 threads per row, float4 sweeps) ----
    {
        const int row = tid & 15;
        const int tq  = tid >> 4;
        float4* Srow4 = (float4*)(S + row * SEQ + tq * 512);
        float lmax = -1e30f;
        for (int i = 0; i < 128; ++i) {
            float4 s = Srow4[i];
            lmax = fmaxf(lmax, fmaxf(fmaxf(s.x, s.y), fmaxf(s.z, s.w)));
        }
        redA[row * 4 + tq] = lmax;
        __syncthreads();
        const float m = fmaxf(fmaxf(redA[row * 4 + 0], redA[row * 4 + 1]),
                              fmaxf(redA[row * 4 + 2], redA[row * 4 + 3]));
        float lsum = 0.0f;
        for (int i = 0; i < 128; ++i) {
            float4 s = Srow4[i];
            s.x = __expf((s.x - m) * 0.125f);   // 1/sqrt(64)
            s.y = __expf((s.y - m) * 0.125f);
            s.z = __expf((s.z - m) * 0.125f);
            s.w = __expf((s.w - m) * 0.125f);
            Srow4[i] = s;
            lsum += (s.x + s.y) + (s.z + s.w);
        }
        redB[row * 4 + tq] = lsum;
        __syncthreads();
        if (tq == 0)
            invR[row] = 1.0f / (redB[row * 4 + 0] + redB[row * 4 + 1] +
                                redB[row * 4 + 2] + redB[row * 4 + 3]);
    }
    __syncthreads();

    // ---- Phase 3: ctx = attn @ V (async V staging, double-buffered) ----
    v8f cc[2];
    for (int j = 0; j < 2; ++j)
        for (int e = 0; e < 8; ++e) cc[j][e] = 0.0f;
    const int dw0 = w * 32;
    const int vr  = tid >> 1;
    const int vco = (tid & 1) * 32;
    const float inv = invR[l16];       // row normalization folded into cvt

    auto stageV = [&](int it, int buf) {
        const unsigned short* gv = V + headBase + (size_t)(it * 32 + vr) * HDIM + vco;
        unsigned la = lds_off(&Vt[buf][vr * HDIM + vco]);
        ASYNC_CP16(la, gv, 0);
        ASYNC_CP16(la, gv, 16);
        ASYNC_CP16(la, gv, 32);
        ASYNC_CP16(la, gv, 48);
    };

    stageV(0, 0);
    ASYNC_WAIT();
    __syncthreads();

    for (int it = 0; it < 64; ++it) {
        const int cur = it & 1;
        const int m0k = it * 32;
        if (it + 1 < 64) stageV(it + 1, cur ^ 1);

        // A fragment: normalized attn weights, f32 -> bf16 on the fly
        FragB16 aS;
#pragma unroll
        for (int v = 0; v < 8; ++v) {
            const int k = half * 16 + 2 * v;
            const float* sp = S + l16 * SEQ + m0k + k;
            aS.u[v] = pk2bf(sp[0] * inv, sp[1] * inv);
        }
        // B fragments from V tile
#pragma unroll
        for (int j = 0; j < 2; ++j) {
            const int d = dw0 + j * 16 + l16;
            FragB16 bV;
#pragma unroll
            for (int v = 0; v < 8; ++v) {
                const int k = half * 16 + 2 * v;
                bV.u[v] = (unsigned int)Vt[cur][k * HDIM + d] |
                          ((unsigned int)Vt[cur][(k + 1) * HDIM + d] << 16);
            }
            cc[j] = wmma_bf16(aS, bV, cc[j]);
        }
        ASYNC_WAIT();
        __syncthreads();
    }

    // store ctx as bf16 [b][ns][h*64+d]
#pragma unroll
    for (int j = 0; j < 2; ++j) {
        const int d = dw0 + j * 16 + l16;
#pragma unroll
        for (int v = 0; v < 8; ++v) {
            const int qr = n0 + half * 8 + v;
            ctx[((size_t)b * SEQ + qr) * EMB + h * HDIM + d] = f2bf(cc[j][v]);
        }
    }
}

// ---------------------------------------------------------------------------
extern "C" void kernel_launch(void* const* d_in, const int* in_sizes, int n_in,
                              void* d_out, int out_size, void* d_ws, size_t ws_size,
                              hipStream_t stream) {
    const float* x  = (const float*)d_in[0];
    const float* Wq = (const float*)d_in[1];
    const float* bq = (const float*)d_in[2];
    const float* Wk = (const float*)d_in[3];
    const float* bk = (const float*)d_in[4];
    const float* Wv = (const float*)d_in[5];
    const float* bv = (const float*)d_in[6];
    const float* Wo = (const float*)d_in[7];
    const float* bo = (const float*)d_in[8];
    float* out = (float*)d_out;

    char* ws = (char*)d_ws;
    const size_t SEG = (size_t)MROWS * EMB * sizeof(unsigned short);  // 8 MB
    unsigned short* xb = (unsigned short*)(ws + 0 * SEG);
    unsigned short* Qb = (unsigned short*)(ws + 1 * SEG);
    unsigned short* Kb = (unsigned short*)(ws + 2 * SEG);
    unsigned short* Vb = (unsigned short*)(ws + 3 * SEG);
    unsigned short* Cb = (unsigned short*)(ws + 4 * SEG);

    const int nElem = MROWS * EMB;  // 4 Mi
    cvt_f32_bf16<<<nElem / (256 * 4), 256, 0, stream>>>(x, xb, nElem);

    dim3 g(MROWS / 128, EMB / 128);   // 32 x 8
    gemm_bf16<0><<<g, 256, 0, stream>>>(xb, Wq, bq, Qb, nullptr);
    gemm_bf16<0><<<g, 256, 0, stream>>>(xb, Wk, bk, Kb, nullptr);
    gemm_bf16<0><<<g, 256, 0, stream>>>(xb, Wv, bv, Vb, nullptr);

    attn_kernel<<<dim3(SEQ / 16, HEADS, 2), 64, 0, stream>>>(Qb, Kb, Vb, Cb);

    gemm_bf16<1><<<g, 256, 0, stream>>>(Cb, Wo, bo, nullptr, out);
}
